// MultiHeadDotProductAttention_14980845928960
// MI455X (gfx1250) — matrix operand order repaired and checked by reading, not compiled, not run on hardware
//
#include <hip/hip_runtime.h>
#include <hip/hip_bf16.h>

// Problem constants (from the reference)
#define BQ   4
#define NBQ  16
#define BSQ  256
#define FQ   1024
#define HQ   16
#define DQ   64
#define HDQ  1024
#define NBLK (BQ * NBQ)   // 64 (b,nb) token blocks of 256 tokens each

typedef __attribute__((ext_vector_type(16))) __bf16 v16bf;
typedef __attribute__((ext_vector_type(8)))  float  v8f;

union FragU { v16bf v; unsigned int u[8]; };

// Native bf16 convert (RTNE) — gfx1250 has hardware bf16 cvt.
__device__ __forceinline__ unsigned short f2bf(float f) {
    union { __bf16 h; unsigned short s; } c;
    c.h = (__bf16)f;
    return c.s;
}

__device__ __forceinline__ float redmax16(float v) {
#pragma unroll
    for (int m = 1; m < 16; m <<= 1) v = fmaxf(v, __shfl_xor(v, m, 32));
    return v;
}
__device__ __forceinline__ float redsum16(float v) {
#pragma unroll
    for (int m = 1; m < 16; m <<= 1) v += __shfl_xor(v, m, 32);
    return v;
}

// Async global -> LDS copy of 16 bytes per lane (ASYNCcnt-tracked, no VGPR data path).
__device__ __forceinline__ void async_ld_b128(unsigned lds_off, unsigned long long gaddr) {
    asm volatile("global_load_async_to_lds_b128 %0, %1, off"
                 :: "v"(lds_off), "v"(gaddr) : "memory");
}
__device__ __forceinline__ void async_ld_b128_o16(unsigned lds_off, unsigned long long gaddr) {
    asm volatile("global_load_async_to_lds_b128 %0, %1, off offset:16"
                 :: "v"(lds_off), "v"(gaddr) : "memory");
}
__device__ __forceinline__ void wait_async0() {
    asm volatile("s_wait_asynccnt 0x0" ::: "memory");
}

#define ASTR 40   // LDS row stride (ushort) for A tile 128x32
#define BSTR 40   // LDS row stride (ushort) for B tile stored transposed 128x32

// One 32-deep WMMA K-step for a 128x128 block tile; wave (wm,wn) owns 64x32.
__device__ __forceinline__ void wmma_tile(const unsigned short* At, const unsigned short* Bt,
                                          v8f acc[4][2], int wm, int wn, int lane) {
    const int ln = lane & 15, hk = lane >> 4;
    FragU a[4], b[2];
#pragma unroll
    for (int mi = 0; mi < 4; ++mi) {
        const unsigned short* ap = At + (wm * 64 + mi * 16 + ln) * ASTR;
#pragma unroll
        for (int v = 0; v < 8; ++v)   // A: K = half*16 + 2v, pairs packed
            a[mi].u[v] = *(const unsigned int*)(ap + hk * 16 + 2 * v);
    }
#pragma unroll
    for (int ni = 0; ni < 2; ++ni) {
        const unsigned short* bp = Bt + (wn * 32 + ni * 16 + ln) * BSTR + hk * 16;
#pragma unroll
        for (int v = 0; v < 8; ++v)   // B: K = half*16 + 2v, pairs packed (Bt is [n][k])
            b[ni].u[v] = *(const unsigned int*)(bp + 2 * v);
    }
#pragma unroll
    for (int mi = 0; mi < 4; ++mi)
#pragma unroll
        for (int ni = 0; ni < 2; ++ni)
            acc[mi][ni] = __builtin_amdgcn_wmma_f32_16x16x32_bf16(
                false, a[mi].v, false, b[ni].v, (short)0, acc[mi][ni], false, false);
}

// ---------------------------------------------------------------------------
// Kernel 1: fused QKV projection + bias + RoPE + q-scale, fp32 -> bf16 out.
// grid = (16 tiles, 64 blks, 3 projections), block = 256 threads (8 waves).
// ---------------------------------------------------------------------------
__global__ __launch_bounds__(256) void qkv_proj_kernel(
    const float* __restrict__ Xq, const float* __restrict__ Xkv,
    const float* __restrict__ Wq, const float* __restrict__ bq,
    const float* __restrict__ Wk, const float* __restrict__ bk,
    const float* __restrict__ Wv, const float* __restrict__ bv,
    unsigned short* __restrict__ qb, unsigned short* __restrict__ kb2,
    unsigned short* __restrict__ vb)
{
    __shared__ unsigned short At[128 * ASTR];
    __shared__ unsigned short Bt[128 * BSTR];

    const int tid = threadIdx.x;
    const int lane = tid & 31, w = tid >> 5;
    const int wm = w >> 2, wn = w & 3;
    const int tile = blockIdx.x, blk = blockIdx.y, z = blockIdx.z;
    const int tm = tile & 1, tn = tile >> 1;

    const float* X    = (z == 0) ? Xq : Xkv;
    const float* W    = (z == 0) ? Wq : ((z == 1) ? Wk : Wv);
    const float* bias = (z == 0) ? bq : ((z == 1) ? bk : bv);
    const float* Ablk = X + (size_t)blk * BSQ * FQ + (size_t)tm * 128 * FQ;

    const v8f z8 = {0.f, 0.f, 0.f, 0.f, 0.f, 0.f, 0.f, 0.f};
    v8f acc[4][2];
#pragma unroll
    for (int i = 0; i < 4; ++i)
#pragma unroll
        for (int j = 0; j < 2; ++j) acc[i][j] = z8;

    for (int k0 = 0; k0 < FQ; k0 += 32) {
        __syncthreads();
        {   // A tile 128x32 fp32 -> bf16 in LDS (conversion in flight: VALU path)
            const int row = tid >> 1, ks = (tid & 1) << 4;
            const float* src = Ablk + (size_t)row * FQ + k0 + ks;
            unsigned short* dst = At + row * ASTR + ks;
#pragma unroll
            for (int j = 0; j < 4; ++j) {
                float4 f = *(const float4*)(src + j * 4);
                ushort4 h;
                h.x = f2bf(f.x); h.y = f2bf(f.y); h.z = f2bf(f.z); h.w = f2bf(f.w);
                *(ushort4*)(dst + j * 4) = h;
            }
        }
        {   // B tile 32x128 fp32 -> bf16, stored transposed [n][k]
            const int kk = tid >> 3, ns = (tid & 7) << 4;
            const float* src = W + (size_t)(k0 + kk) * HDQ + tn * 128 + ns;
#pragma unroll
            for (int j = 0; j < 4; ++j) {
                float4 f = *(const float4*)(src + j * 4);
                Bt[(ns + j * 4 + 0) * BSTR + kk] = f2bf(f.x);
                Bt[(ns + j * 4 + 1) * BSTR + kk] = f2bf(f.y);
                Bt[(ns + j * 4 + 2) * BSTR + kk] = f2bf(f.z);
                Bt[(ns + j * 4 + 3) * BSTR + kk] = f2bf(f.w);
            }
        }
        __syncthreads();
        if (k0 + 32 < FQ)
            __builtin_prefetch(Ablk + (size_t)(tid >> 1) * FQ + k0 + 32, 0, 1);
        wmma_tile(At, Bt, acc, wm, wn, lane);
    }

    // Epilogue: bias, RoPE on first 32 dims of each head, q *= 1/sqrt(D)
    const int ln = lane & 15, hk = lane >> 4;
#pragma unroll
    for (int mi = 0; mi < 4; ++mi) {
#pragma unroll
        for (int ni = 0; ni < 2; ++ni) {
            const int col = tn * 128 + wn * 32 + ni * 16 + ln;   // = h*64 + d
            const int h = col >> 6, d = col & 63;
            const float bval = bias[col];
            const int rbase = tm * 128 + wm * 64 + mi * 16 + hk * 8;
#pragma unroll
            for (int r = 0; r < 8; ++r) {
                float val = acc[mi][ni][r] + bval;
                const int s = rbase + r;
                if (z < 2) {
                    const float part = __shfl_xor(val, 1, 32);   // pair lives in adjacent lane
                    if (d < 32) {
                        const int nb = blk & (NBQ - 1);
                        const float pos  = (float)(nb * BSQ + s);
                        const float freq = __expf(-(float)(d & ~1) * (9.210340372f / 32.0f));
                        const float ang  = pos * freq;
                        const float rot  = (d & 1) ? part : -part;
                        val = val * __cosf(ang) + rot * __sinf(ang);
                    }
                    if (z == 0) val *= 0.125f;                   // 1/sqrt(64)
                    unsigned short* dst = (z == 0) ? qb : kb2;
                    dst[(((size_t)blk * HQ + h) * BSQ + s) * DQ + d] = f2bf(val);
                } else {
                    // V stored transposed per head: [h][d][s]
                    vb[(((size_t)blk * HQ + h) * DQ + d) * BSQ + s] = f2bf(val);
                }
            }
        }
    }
}

// ---------------------------------------------------------------------------
// Kernel 2: flash-style attention, one block per (blk, head).
// 8 waves x 32 query rows; 4 key chunks of 64; WMMA for S=QK^T and O+=PV.
// K/V chunks arrive via async global->LDS DMA (ASYNCcnt), no VGPR staging.
// ---------------------------------------------------------------------------
__global__ __launch_bounds__(256) void attn_kernel(
    const unsigned short* __restrict__ qb, const unsigned short* __restrict__ kbuf,
    const unsigned short* __restrict__ vbuf, unsigned short* __restrict__ xb)
{
    __shared__ unsigned short kt[64 * 64];       // K chunk [key][d]
    __shared__ unsigned short vt[64 * 64];       // V chunk [d][key]
    __shared__ unsigned short pb[8 * 32 * 64];   // per-wave P scratch

    const int bh = blockIdx.x;
    const int blk = bh >> 4, h = bh & 15;
    const unsigned short* qh = qb   + (size_t)bh * BSQ * DQ;
    const unsigned short* kh = kbuf + (size_t)bh * BSQ * DQ;
    const unsigned short* vh = vbuf + (size_t)bh * DQ * BSQ;

    const int tid = threadIdx.x;
    const int lane = tid & 31, w = tid >> 5;
    const int ln = lane & 15, hk = lane >> 4;

    // Resident Q A-frags: 32 rows x 64 K (2 M-frags x 2 K-frags)
    FragU aq[2][2];
#pragma unroll
    for (int mi = 0; mi < 2; ++mi)
#pragma unroll
        for (int kf = 0; kf < 2; ++kf) {
            const unsigned short* qp = qh + (size_t)(w * 32 + mi * 16 + ln) * DQ + kf * 32;
#pragma unroll
            for (int v = 0; v < 8; ++v)
                aq[mi][kf].u[v] = *(const unsigned int*)(qp + hk * 16 + 2 * v);
        }

    float m[2][8], l[2][8];
    const v8f z8 = {0.f, 0.f, 0.f, 0.f, 0.f, 0.f, 0.f, 0.f};
    v8f o[2][4];
#pragma unroll
    for (int mi = 0; mi < 2; ++mi) {
#pragma unroll
        for (int r = 0; r < 8; ++r) { m[mi][r] = -1e30f; l[mi][r] = 0.f; }
#pragma unroll
        for (int nd = 0; nd < 4; ++nd) o[mi][nd] = z8;
    }

    unsigned short* pw = pb + w * 32 * 64;

    for (int kc = 0; kc < 4; ++kc) {
        __syncthreads();
        {   // async DMA both 8KB chunks: 512 16-byte segments each, 2 per thread
#pragma unroll
            for (int ss = tid; ss < 512; ss += 256) {
                const int row = ss >> 3, cs = (ss & 7) << 3;     // 8 ushort = 16B
                const unsigned kdst = (unsigned)(size_t)(kt + row * 64 + cs);
                const unsigned long long ksrc =
                    (unsigned long long)(size_t)(kh + (size_t)(kc * 64 + row) * DQ + cs);
                async_ld_b128(kdst, ksrc);
                const unsigned vdst2 = (unsigned)(size_t)(vt + row * 64 + cs);
                const unsigned long long vsrc =
                    (unsigned long long)(size_t)(vh + (size_t)row * BSQ + kc * 64 + cs);
                async_ld_b128(vdst2, vsrc);
            }
            wait_async0();
        }
        __syncthreads();

        // S = Q x K^T  (32 x 64 per wave)
        v8f s[2][4];
#pragma unroll
        for (int mi = 0; mi < 2; ++mi)
#pragma unroll
            for (int ni = 0; ni < 4; ++ni) s[mi][ni] = z8;
#pragma unroll
        for (int kf = 0; kf < 2; ++kf) {
            FragU bk[4];
#pragma unroll
            for (int ni = 0; ni < 4; ++ni) {
                const unsigned short* bp = kt + (ni * 16 + ln) * 64 + kf * 32 + hk * 16;
#pragma unroll
                for (int v = 0; v < 8; ++v)
                    bk[ni].u[v] = *(const unsigned int*)(bp + 2 * v);
            }
#pragma unroll
            for (int mi = 0; mi < 2; ++mi)
#pragma unroll
                for (int ni = 0; ni < 4; ++ni)
                    s[mi][ni] = __builtin_amdgcn_wmma_f32_16x16x32_bf16(
                        false, aq[mi][kf].v, false, bk[ni].v, (short)0, s[mi][ni], false, false);
        }

        // Online softmax update + P -> LDS (bf16)
#pragma unroll
        for (int mi = 0; mi < 2; ++mi) {
#pragma unroll
            for (int r = 0; r < 8; ++r) {
                float cm = s[mi][0][r];
#pragma unroll
                for (int ni = 1; ni < 4; ++ni) cm = fmaxf(cm, s[mi][ni][r]);
                cm = redmax16(cm);
                const float mo = m[mi][r];
                const float mn = fmaxf(mo, cm);
                const float sc = __expf(mo - mn);
                m[mi][r] = mn;
                float rs = 0.f;
                const int prow = mi * 16 + hk * 8 + r;
#pragma unroll
                for (int ni = 0; ni < 4; ++ni) {
                    const float p = __expf(s[mi][ni][r] - mn);
                    rs += p;
                    pw[prow * 64 + ni * 16 + ln] = f2bf(p);
                }
                rs = redsum16(rs);
                l[mi][r] = l[mi][r] * sc + rs;
#pragma unroll
                for (int nd = 0; nd < 4; ++nd) o[mi][nd][r] *= sc;
            }
        }

        // O += P x V   (per-wave LDS is RAW-safe: LDS ops are in-order per wave)
#pragma unroll
        for (int kf = 0; kf < 2; ++kf) {
            FragU ap[2], bv[4];
#pragma unroll
            for (int mi = 0; mi < 2; ++mi) {
                const unsigned short* app = pw + (mi * 16 + ln) * 64 + kf * 32;
#pragma unroll
                for (int v = 0; v < 8; ++v)
                    ap[mi].u[v] = *(const unsigned int*)(app + hk * 16 + 2 * v);
            }
#pragma unroll
            for (int nd = 0; nd < 4; ++nd) {
                const unsigned short* bp = vt + (nd * 16 + ln) * 64 + kf * 32 + hk * 16;
#pragma unroll
                for (int v = 0; v < 8; ++v)
                    bv[nd].u[v] = *(const unsigned int*)(bp + 2 * v);
            }
#pragma unroll
            for (int mi = 0; mi < 2; ++mi)
#pragma unroll
                for (int nd = 0; nd < 4; ++nd)
                    o[mi][nd] = __builtin_amdgcn_wmma_f32_16x16x32_bf16(
                        false, ap[mi].v, false, bv[nd].v, (short)0, o[mi][nd], false, false);
        }
    }

    // Normalize and write x[b,nb,s, h*64+d] as bf16 for the output projection
#pragma unroll
    for (int mi = 0; mi < 2; ++mi)
#pragma unroll
        for (int nd = 0; nd < 4; ++nd)
#pragma unroll
            for (int r = 0; r < 8; ++r) {
                const int srow = w * 32 + mi * 16 + hk * 8 + r;
                const float val = o[mi][nd][r] / l[mi][r];
                xb[((size_t)blk * BSQ + srow) * HDQ + h * DQ + nd * 16 + ln] = f2bf(val);
            }
}

// ---------------------------------------------------------------------------
// Kernel 3: output projection, bf16 A from workspace (async DMA to LDS),
// fp32 Wo -> fp32 out.
// ---------------------------------------------------------------------------
__global__ __launch_bounds__(256) void out_proj_kernel(
    const unsigned short* __restrict__ Xb, const float* __restrict__ Wo,
    const float* __restrict__ bo, float* __restrict__ out)
{
    __shared__ unsigned short At[128 * ASTR];
    __shared__ unsigned short Bt[128 * BSTR];

    const int tid = threadIdx.x;
    const int lane = tid & 31, w = tid >> 5;
    const int wm = w >> 2, wn = w & 3;
    const int tile = blockIdx.x, blk = blockIdx.y;
    const int tm = tile & 1, tn = tile >> 1;
    const unsigned short* Ablk = Xb + (size_t)blk * BSQ * HDQ + (size_t)tm * 128 * HDQ;

    const v8f z8 = {0.f, 0.f, 0.f, 0.f, 0.f, 0.f, 0.f, 0.f};
    v8f acc[4][2];
#pragma unroll
    for (int i = 0; i < 4; ++i)
#pragma unroll
        for (int j = 0; j < 2; ++j) acc[i][j] = z8;

    for (int k0 = 0; k0 < HDQ; k0 += 32) {
        __syncthreads();
        {   // A tile already bf16: async global->LDS DMA, 32B per thread
            const int row = tid >> 1, ks = (tid & 1) << 4;
            const unsigned dst = (unsigned)(size_t)(At + row * ASTR + ks);
            const unsigned long long src =
                (unsigned long long)(size_t)(Ablk + (size_t)row * HDQ + k0 + ks);
            async_ld_b128(dst, src);
            async_ld_b128_o16(dst, src);
        }
        {   // B tile: Wo is [h*64+d][f] fp32; convert + store transposed bf16 [n][k]
            const int kk = tid >> 3, ns = (tid & 7) << 4;
            const float* src = Wo + (size_t)(k0 + kk) * FQ + tn * 128 + ns;
#pragma unroll
            for (int j = 0; j < 4; ++j) {
                float4 f = *(const float4*)(src + j * 4);
                Bt[(ns + j * 4 + 0) * BSTR + kk] = f2bf(f.x);
                Bt[(ns + j * 4 + 1) * BSTR + kk] = f2bf(f.y);
                Bt[(ns + j * 4 + 2) * BSTR + kk] = f2bf(f.z);
                Bt[(ns + j * 4 + 3) * BSTR + kk] = f2bf(f.w);
            }
        }
        wait_async0();
        __syncthreads();
        if (k0 + 32 < HDQ)
            __builtin_prefetch(Ablk + (size_t)(tid >> 1) * HDQ + k0 + 32, 0, 1);
        wmma_tile(At, Bt, acc, wm, wn, lane);
    }

    const int ln = lane & 15, hk = lane >> 4;
#pragma unroll
    for (int mi = 0; mi < 4; ++mi)
#pragma unroll
        for (int ni = 0; ni < 2; ++ni) {
            const int col = tn * 128 + wn * 32 + ni * 16 + ln;
            const float bval = bo[col];
            const int rbase = tm * 128 + wm * 64 + mi * 16 + hk * 8;
#pragma unroll
            for (int r = 0; r < 8; ++r)
                out[((size_t)blk * BSQ + (rbase + r)) * FQ + col] = acc[mi][ni][r] + bval;
        }
}

// ---------------------------------------------------------------------------
extern "C" void kernel_launch(void* const* d_in, const int* in_sizes, int n_in,
                              void* d_out, int out_size, void* d_ws, size_t ws_size,
                              hipStream_t stream) {
    (void)in_sizes; (void)n_in; (void)out_size; (void)ws_size;
    const float* Xq  = (const float*)d_in[0];
    const float* Xkv = (const float*)d_in[1];
    const float* Wq  = (const float*)d_in[2];
    const float* bq  = (const float*)d_in[3];
    const float* Wk  = (const float*)d_in[4];
    const float* bk  = (const float*)d_in[5];
    const float* Wv  = (const float*)d_in[6];
    const float* bv  = (const float*)d_in[7];
    const float* Wo  = (const float*)d_in[8];
    const float* bo  = (const float*)d_in[9];
    float* out = (float*)d_out;

    const size_t HDTOT = (size_t)NBLK * BSQ * HDQ;   // 16,777,216 elements
    unsigned short* qb = (unsigned short*)d_ws;      // 32 MB each
    unsigned short* kb = qb + HDTOT;
    unsigned short* vb = kb + HDTOT;
    unsigned short* xb = vb + HDTOT;

    dim3 gProj(16, NBLK, 3);
    qkv_proj_kernel<<<gProj, 256, 0, stream>>>(Xq, Xkv, Wq, bq, Wk, bk, Wv, bv, qb, kb, vb);
    attn_kernel<<<NBLK * HQ, 256, 0, stream>>>(qb, kb, vb, xb);
    dim3 gOut(16, NBLK);
    out_proj_kernel<<<gOut, 256, 0, stream>>>(xb, Wo, bo, out);
}
